// Project_Transformer_49572512530942
// MI455X (gfx1250) — matrix operand order repaired and checked by reading, not compiled, not run on hardware
//
#include <hip/hip_runtime.h>
#include <math.h>

// Problem constants (match reference)
#define BB 4
#define NN 2048
#define CC 128
#define DD 128
#define KKN 16          // kNN neighbors
#define HHID 64         // xyz/pos MLP hidden
#define AATT 512        // attention MLP hidden
#define BN (BB*NN)      // 8192 columns for 1x1 convs
#define PAIRS (BN*KKN)  // 131072 (b,n,k) pairs

typedef __attribute__((ext_vector_type(16))) __bf16 v16bf;
typedef __attribute__((ext_vector_type(8)))  float  v8f;
typedef unsigned short bfraw;

// ---------------- bf16 helpers ----------------
__device__ __forceinline__ bfraw f2bf(float f) {
  union { float f; unsigned u; } v; v.f = f;
  unsigned r = v.u + 0x7FFFu + ((v.u >> 16) & 1u);
  return (bfraw)(r >> 16);
}
__device__ __forceinline__ float bf2f(bfraw b) {
  union { unsigned u; float f; } v; v.u = ((unsigned)b) << 16; return v.f;
}

union FragU { v16bf v; float4 f4[2]; };

// A fragment, 16x32 bf16 (row-major source, ld = K stride).
__device__ __forceinline__ v16bf load_afrag(const bfraw* __restrict__ W,
                                            int row, int ld, int k0,
                                            int lo, int hi) {
  const bfraw* p = W + (size_t)(row + lo) * ld + k0 + hi * 8;
  FragU f;
  f.f4[0] = *reinterpret_cast<const float4*>(p);
  f.f4[1] = *reinterpret_cast<const float4*>(p + 16);
  return f.v;
}
// B fragment, 32x16 bf16 from a column-major source (K contiguous per column).
__device__ __forceinline__ v16bf load_bfrag(const bfraw* __restrict__ colptr,
                                            int k0, int hi) {
  const bfraw* p = colptr + k0 + hi * 8;
  FragU f;
  f.f4[0] = *reinterpret_cast<const float4*>(p);
  f.f4[1] = *reinterpret_cast<const float4*>(p + 16);
  return f.v;
}

__device__ __forceinline__ v8f wmma_bf16(v16bf a, v16bf b, v8f c) {
  return __builtin_amdgcn_wmma_f32_16x16x32_bf16(false, a, false, b,
                                                 (short)0, c, false, false);
}

// ---------------- conversion kernels ----------------
__global__ void wcvt_kernel(const float* __restrict__ s, bfraw* __restrict__ d, int n) {
  int i = blockIdx.x * blockDim.x + threadIdx.x;
  if (i < n) d[i] = f2bf(s[i]);
}

// (B, Ctot, N) f32 (row-major input) -> X[col=(b*N+n)][coff + c] bf16
__global__ void xcvt_kernel(const float* __restrict__ src, bfraw* __restrict__ X,
                            int Ctot, int ldx, int coff) {
  size_t i = (size_t)blockIdx.x * blockDim.x + threadIdx.x;
  size_t total = (size_t)BB * Ctot * NN;
  if (i >= total) return;
  int n = (int)(i % NN);
  size_t t = i / NN;
  int c = (int)(t % Ctot);
  int b = (int)(t / Ctot);
  X[((size_t)(b * NN + n)) * ldx + coff + c] = f2bf(src[i]);
}

// ---------------- generic WMMA GEMM (M = 128) ----------------
// Activations are column-major [col][c] bf16.  out: col-major, or row-major
// (B,128,N) for the final kernel.  res (optional) always col-major.
template <int KD>
__global__ __launch_bounds__(256) void gemm_bf16_kernel(
    const bfraw* __restrict__ W, const bfraw* __restrict__ X,
    const float* __restrict__ bias, const float* __restrict__ res,
    float* __restrict__ out, int ldx, int xoff, int relu, int rowmajor) {
  int lane = threadIdx.x & 31, wave = threadIdx.x >> 5;
  int lo = lane & 15, hi = lane >> 4;
  int colbase = blockIdx.x * 128;
  int rowb = wave * 16;

  // Hoist the wave's entire A slice (16 rows x KD) into registers.
  v16bf aw[KD / 32];
#pragma unroll
  for (int kk = 0; kk < KD / 32; ++kk)
    aw[kk] = load_afrag(W, rowb, KD, kk * 32, lo, hi);

  v8f acc[8];
#pragma unroll
  for (int i = 0; i < 8; ++i) { v8f z = {0,0,0,0,0,0,0,0}; acc[i] = z; }

#pragma unroll
  for (int kk = 0; kk < KD / 32; ++kk) {
    v16bf bfr[8];
#pragma unroll
    for (int ct = 0; ct < 8; ++ct)
      bfr[ct] = load_bfrag(X + (size_t)(colbase + ct * 16 + lo) * ldx + xoff,
                           kk * 32, hi);
#pragma unroll
    for (int ct = 0; ct < 8; ++ct)
      acc[ct] = wmma_bf16(aw[kk], bfr[ct], acc[ct]);
  }

  int mrow = rowb + hi * 8;
#pragma unroll
  for (int ct = 0; ct < 8; ++ct) {
    int col = colbase + ct * 16 + lo;
    float vv[8];
#pragma unroll
    for (int r = 0; r < 8; ++r) {
      float v = acc[ct][r] + bias[mrow + r];
      if (relu) v = fmaxf(v, 0.0f);
      vv[r] = v;
    }
    if (res) {
      const float* pr = res + (size_t)col * DD + mrow;
      float4 r0 = *reinterpret_cast<const float4*>(pr);
      float4 r1 = *reinterpret_cast<const float4*>(pr + 4);
      vv[0] += r0.x; vv[1] += r0.y; vv[2] += r0.z; vv[3] += r0.w;
      vv[4] += r1.x; vv[5] += r1.y; vv[6] += r1.z; vv[7] += r1.w;
    }
    if (rowmajor) {
      int bidx = col >> 11;     // / 2048
      int n    = col & 2047;
      float* po = out + ((size_t)bidx * DD + mrow) * NN + n;
#pragma unroll
      for (int r = 0; r < 8; ++r) po[(size_t)r * NN] = vv[r];
    } else {
      float* po = out + (size_t)col * DD + mrow;
      *reinterpret_cast<float4*>(po)     = make_float4(vv[0], vv[1], vv[2], vv[3]);
      *reinterpret_cast<float4*>(po + 4) = make_float4(vv[4], vv[5], vv[6], vv[7]);
    }
  }
}

// ---------------- kNN top-16 (include self) ----------------
__global__ void knn_kernel(const float* __restrict__ coord, int dim,
                           int* __restrict__ idx) {
  int t = blockIdx.x * blockDim.x + threadIdx.x;
  if (t >= BN) return;
  int b = t / NN, n = t % NN;
  const float* cb = coord + (size_t)b * dim * NN;
  float q0 = cb[n], q1 = cb[NN + n];
  float q2 = (dim == 3) ? cb[2 * NN + n] : 0.0f;
  float bd[KKN]; int bi[KKN];
#pragma unroll
  for (int j = 0; j < KKN; ++j) { bd[j] = 3.4e38f; bi[j] = 0; }
  for (int m = 0; m < NN; ++m) {
    float d0 = cb[m] - q0, d1 = cb[NN + m] - q1;
    float d2 = d0 * d0 + d1 * d1;
    if (dim == 3) { float dz = cb[2 * NN + m] - q2; d2 += dz * dz; }
    if (d2 < bd[KKN - 1]) {
      float cd = d2; int ci = m;
#pragma unroll
      for (int j = 0; j < KKN; ++j) {
        if (cd < bd[j]) {
          float td = bd[j]; int ti = bi[j];
          bd[j] = cd; bi[j] = ci; cd = td; ci = ti;
        }
      }
    }
  }
#pragma unroll
  for (int j = 0; j < KKN; ++j) idx[(size_t)t * KKN + j] = bi[j];
}

// ---------------- per-pair coordinate MLP + T/EV build ----------------
// fa/fb/fv are column-major [bn][d].  T/EV written pair-major, d contiguous.
__global__ __launch_bounds__(256) void emb_kernel(
    const float* __restrict__ coord, int dim, const int* __restrict__ idx,
    const float* __restrict__ fa, const float* __restrict__ fb,
    const float* __restrict__ fv,
    const float* __restrict__ W1, const float* __restrict__ B1,
    const float* __restrict__ W2, const float* __restrict__ B2,
    bfraw* __restrict__ T, bfraw* __restrict__ EV) {
  __shared__ float sW2[DD * HHID];
  __shared__ float sW1[HHID * 3];
  __shared__ float sB1[HHID];
  __shared__ float sB2[DD];
  for (int i = threadIdx.x; i < DD * HHID; i += blockDim.x) sW2[i] = W2[i];
  for (int i = threadIdx.x; i < HHID * dim; i += blockDim.x) sW1[i] = W1[i];
  for (int i = threadIdx.x; i < HHID; i += blockDim.x) sB1[i] = B1[i];
  for (int i = threadIdx.x; i < DD; i += blockDim.x) sB2[i] = B2[i];
  __syncthreads();
  int pair = blockIdx.x * blockDim.x + threadIdx.x;
  if (pair >= PAIRS) return;
  int bn = pair / KKN;
  int b = bn / NN, n = bn % NN;
  int m = idx[pair];
  const float* cb = coord + (size_t)b * dim * NN;
  float rel[3] = {0.f, 0.f, 0.f};
  for (int j = 0; j < dim; ++j) rel[j] = cb[j * NN + n] - cb[j * NN + m];
  float h[HHID];
#pragma unroll
  for (int t = 0; t < HHID; ++t) {
    float s = sB1[t];
    for (int j = 0; j < dim; ++j) s += sW1[t * dim + j] * rel[j];
    h[t] = fmaxf(s, 0.0f);
  }
  const float* pfa = fa + (size_t)bn * DD;
  const float* pfb = fb + ((size_t)(b * NN + m)) * DD;
  const float* pfv = fv + (size_t)bn * DD;
  bfraw* pT = T + (size_t)pair * DD;
  bfraw* pE = EV + (size_t)pair * DD;
  for (int d8 = 0; d8 < DD; d8 += 8) {
    union { float4 f; bfraw u[8]; } pkT, pkE;
#pragma unroll
    for (int j = 0; j < 8; ++j) {
      int d = d8 + j;
      float e = sB2[d];
      const float* wr = &sW2[d * HHID];
#pragma unroll
      for (int t = 0; t < HHID; ++t) e += wr[t] * h[t];
      pkT.u[j] = f2bf(pfa[d] - pfb[d] + e);
      pkE.u[j] = f2bf(pfv[d] + e);
    }
    *reinterpret_cast<float4*>(&pT[d8]) = pkT.f;
    *reinterpret_cast<float4*>(&pE[d8]) = pkE.f;
  }
}

// ---------------- fused attention MLP (the WMMA workhorse) ----------------
#define LDH 520   // 512 + 8 pad: stride 1040B -> +4 banks per column, no conflicts
__global__ __launch_bounds__(256) void attn_kernel(
    const bfraw* __restrict__ T, const bfraw* __restrict__ EV,
    const bfraw* __restrict__ W1, const float* __restrict__ B1,
    const bfraw* __restrict__ W2, const float* __restrict__ B2,
    float* __restrict__ agg) {
  extern __shared__ bfraw hid[];  // [128 pairs][LDH]
  int lane = threadIdx.x & 31, wave = threadIdx.x >> 5;
  int lo = lane & 15, hi = lane >> 4;
  size_t pairbase = (size_t)blockIdx.x * 128;

  // ---- GEMM1: wave owns 64 hidden rows; W1 slice lives in VGPRs (128) ----
  int arow = wave * 64;
  v16bf aw1[16];  // [kk][rt]
#pragma unroll
  for (int kk = 0; kk < 4; ++kk)
#pragma unroll
    for (int rt = 0; rt < 4; ++rt)
      aw1[kk * 4 + rt] = load_afrag(W1, arow + rt * 16, DD, kk * 32, lo, hi);

  for (int ct = 0; ct < 8; ++ct) {
    int col = ct * 16 + lo;
    const bfraw* tcol = T + (pairbase + col) * DD;
    v16bf bfr[4];
#pragma unroll
    for (int kk = 0; kk < 4; ++kk) bfr[kk] = load_bfrag(tcol, kk * 32, hi);
    v8f acc[4];
#pragma unroll
    for (int i = 0; i < 4; ++i) { v8f z = {0,0,0,0,0,0,0,0}; acc[i] = z; }
#pragma unroll
    for (int kk = 0; kk < 4; ++kk)
#pragma unroll
      for (int rt = 0; rt < 4; ++rt)
        acc[rt] = wmma_bf16(aw1[kk * 4 + rt], bfr[kk], acc[rt]);
#pragma unroll
    for (int rt = 0; rt < 4; ++rt) {
      int ab = arow + rt * 16 + hi * 8;
      union { float4 f; bfraw u[8]; } pk;
#pragma unroll
      for (int r = 0; r < 8; ++r)
        pk.u[r] = f2bf(fmaxf(acc[rt][r] + B1[ab + r], 0.0f));
      *reinterpret_cast<float4*>(&hid[(size_t)col * LDH + ab]) = pk.f;
    }
  }
  __syncthreads();

  // ---- GEMM2: wave owns 16 D-rows x all 128 cols ----
  int drow = wave * 16;
  v8f acc2[8];
#pragma unroll
  for (int i = 0; i < 8; ++i) { v8f z = {0,0,0,0,0,0,0,0}; acc2[i] = z; }
  for (int k0 = 0; k0 < AATT; k0 += 32) {
    v16bf af = load_afrag(W2, drow, AATT, k0, lo, hi);
    v16bf bfr[8];
#pragma unroll
    for (int ct = 0; ct < 8; ++ct)
      bfr[ct] = load_bfrag(&hid[(size_t)(ct * 16 + lo) * LDH], k0, hi);
#pragma unroll
    for (int ct = 0; ct < 8; ++ct)
      acc2[ct] = wmma_bf16(af, bfr[ct], acc2[ct]);
  }

  // ---- epilogue: bias + softmax_one over K (16 lanes) + aggregate ----
  int dbase = drow + hi * 8;
  float bs[8];
#pragma unroll
  for (int r = 0; r < 8; ++r) bs[r] = B2[dbase + r];
#pragma unroll
  for (int g = 0; g < 8; ++g) {    // col tile g == one point n, 16 neighbors
    size_t pair = pairbase + (size_t)g * 16 + lo;
    const bfraw* evp = EV + pair * DD + dbase;
    float outv[8];
#pragma unroll
    for (int r = 0; r < 8; ++r) {
      float x = acc2[g][r] + bs[r];
      float mx = x;
      mx = fmaxf(mx, __shfl_xor(mx, 1, 16));
      mx = fmaxf(mx, __shfl_xor(mx, 2, 16));
      mx = fmaxf(mx, __shfl_xor(mx, 4, 16));
      mx = fmaxf(mx, __shfl_xor(mx, 8, 16));
      float e = __expf(x - mx);
      float s = e;
      s += __shfl_xor(s, 1, 16);
      s += __shfl_xor(s, 2, 16);
      s += __shfl_xor(s, 4, 16);
      s += __shfl_xor(s, 8, 16);
      float xa = e / (1.0f + s);           // softmax_one
      float p = xa * bf2f(evp[r]);
      p += __shfl_xor(p, 1, 16);
      p += __shfl_xor(p, 2, 16);
      p += __shfl_xor(p, 4, 16);
      p += __shfl_xor(p, 8, 16);
      outv[r] = p;
    }
    if (lo == 0) {
      size_t bn = (pairbase + (size_t)g * 16) / KKN;  // == column index
      float* po = agg + bn * DD + dbase;              // col-major agg
      *reinterpret_cast<float4*>(po) =
          make_float4(outv[0], outv[1], outv[2], outv[3]);
      *reinterpret_cast<float4*>(po + 4) =
          make_float4(outv[4], outv[5], outv[6], outv[7]);
    }
  }
}

// ---------------- host orchestration ----------------
extern "C" void kernel_launch(void* const* d_in, const int* in_sizes, int n_in,
                              void* d_out, int out_size, void* d_ws, size_t ws_size,
                              hipStream_t stream) {
  const float* xyz    = (const float*)d_in[0];
  const float* pos    = (const float*)d_in[1];
  const float* key    = (const float*)d_in[2];
  const float* query  = (const float*)d_in[3];
  const float* mv_w1  = (const float*)d_in[4];  const float* mv_b1 = (const float*)d_in[5];
  const float* mv_w2  = (const float*)d_in[6];  const float* mv_b2 = (const float*)d_in[7];
  const float* mv_ws  = (const float*)d_in[8];  const float* mv_bs = (const float*)d_in[9];
  const float* wk     = (const float*)d_in[10]; const float* bk    = (const float*)d_in[11];
  const float* wq     = (const float*)d_in[12]; const float* bq    = (const float*)d_in[13];
  const float* wv     = (const float*)d_in[14]; const float* bv    = (const float*)d_in[15];
  const float* xyz_w1 = (const float*)d_in[16]; const float* xyz_b1= (const float*)d_in[17];
  const float* xyz_w2 = (const float*)d_in[18]; const float* xyz_b2= (const float*)d_in[19];
  const float* pos_w1 = (const float*)d_in[20]; const float* pos_b1= (const float*)d_in[21];
  const float* pos_w2 = (const float*)d_in[22]; const float* pos_b2= (const float*)d_in[23];
  const float* xat_w1 = (const float*)d_in[24]; const float* xat_b1= (const float*)d_in[25];
  const float* xat_w2 = (const float*)d_in[26]; const float* xat_b2= (const float*)d_in[27];
  const float* at_w1  = (const float*)d_in[28]; const float* at_b1 = (const float*)d_in[29];
  const float* at_w2  = (const float*)d_in[30]; const float* at_b2 = (const float*)d_in[31];
  const float* wkq    = (const float*)d_in[32]; const float* bkq   = (const float*)d_in[33];
  const float* wend   = (const float*)d_in[34]; const float* bend  = (const float*)d_in[35];
  float* out = (float*)d_out;

  // bump allocator in workspace (needs ~110 MB)
  char* base = (char*)d_ws;
  size_t off = 0;
  auto alloc = [&](size_t bytes) -> void* {
    void* p = base + off;
    off += (bytes + 255) & ~(size_t)255;
    return p;
  };

  bfraw* wb_mv1 = (bfraw*)alloc(128 * 256 * 2);
  bfraw* wb_mvs = (bfraw*)alloc(128 * 256 * 2);
  bfraw* wb_mv2 = (bfraw*)alloc(128 * 128 * 2);
  bfraw* wb_wk  = (bfraw*)alloc(128 * 128 * 2);
  bfraw* wb_wq  = (bfraw*)alloc(128 * 128 * 2);
  bfraw* wb_wv  = (bfraw*)alloc(128 * 128 * 2);
  bfraw* wb_wkq = (bfraw*)alloc(128 * 128 * 2);
  bfraw* wb_wend= (bfraw*)alloc(128 * 128 * 2);
  bfraw* wb_x1  = (bfraw*)alloc(512 * 128 * 2);
  bfraw* wb_x2  = (bfraw*)alloc(128 * 512 * 2);
  bfraw* wb_a1  = (bfraw*)alloc(512 * 128 * 2);
  bfraw* wb_a2  = (bfraw*)alloc(128 * 512 * 2);
  bfraw* Xcat   = (bfraw*)alloc((size_t)BN * 256 * 2);
  bfraw* Xa     = (bfraw*)alloc((size_t)BN * 128 * 2);
  float* val0   = (float*)alloc((size_t)BN * 128 * 4);   // identity (col-major)
  float* hmid   = (float*)alloc((size_t)BN * 128 * 4);
  float* kbuf   = (float*)alloc((size_t)BN * 128 * 4);
  float* qbuf   = (float*)alloc((size_t)BN * 128 * 4);
  float* vbuf   = (float*)alloc((size_t)BN * 128 * 4);
  float* tmp1   = (float*)alloc((size_t)BN * 128 * 4);
  float* val2   = (float*)alloc((size_t)BN * 128 * 4);
  float* aggb   = (float*)alloc((size_t)BN * 128 * 4);
  int*   idx1   = (int*)alloc((size_t)PAIRS * 4);
  int*   idx2   = (int*)alloc((size_t)PAIRS * 4);
  bfraw* Tbuf   = (bfraw*)alloc((size_t)PAIRS * 128 * 2);
  bfraw* EVbuf  = (bfraw*)alloc((size_t)PAIRS * 128 * 2);
  (void)ws_size; (void)in_sizes; (void)n_in; (void)out_size;

  auto wc = [&](const float* s, bfraw* d, int n) {
    wcvt_kernel<<<(n + 255) / 256, 256, 0, stream>>>(s, d, n);
  };
  // transpose-convert a row-major (B,128,N) input into Xcat section
  auto xc_t = [&](const float* s, bfraw* d, int ldx, int coff) {
    int total = BB * 128 * NN;
    xcvt_kernel<<<(total + 255) / 256, 256, 0, stream>>>(s, d, 128, ldx, coff);
  };
  // flat convert a col-major f32 activation -> col-major bf16
  auto xc_cm = [&](const float* s, bfraw* d) { wc(s, d, BN * 128); };

  auto gemm128 = [&](const bfraw* W, const bfraw* X, const float* bias,
                     const float* res, float* o, int ldx, int xoff,
                     int relu, int rowmajor) {
    gemm_bf16_kernel<128><<<BN / 128, 256, 0, stream>>>(W, X, bias, res, o,
                                                        ldx, xoff, relu, rowmajor);
  };
  auto gemm256 = [&](const bfraw* W, const bfraw* X, const float* bias,
                     const float* res, float* o, int ldx, int xoff,
                     int relu, int rowmajor) {
    gemm_bf16_kernel<256><<<BN / 128, 256, 0, stream>>>(W, X, bias, res, o,
                                                        ldx, xoff, relu, rowmajor);
  };

  // weight conversions
  wc(mv_w1, wb_mv1, 128 * 256);
  wc(mv_ws, wb_mvs, 128 * 256);
  wc(mv_w2, wb_mv2, 128 * 128);
  wc(wk, wb_wk, 128 * 128);
  wc(wq, wb_wq, 128 * 128);
  wc(wv, wb_wv, 128 * 128);
  wc(wkq, wb_wkq, 128 * 128);
  wc(wend, wb_wend, 128 * 128);
  wc(xat_w1, wb_x1, 512 * 128);
  wc(xat_w2, wb_x2, 128 * 512);
  wc(at_w1, wb_a1, 512 * 128);
  wc(at_w2, wb_a2, 128 * 512);

  // cat(key, query) column-major bf16 (ldx = 256)
  xc_t(key, Xcat, 256, 0);
  xc_t(query, Xcat, 256, 128);

  // MLP_Res: val0 = mv_w2·relu(mv_w1·cat+b1)+b2 + (mv_ws·cat+bs)
  gemm256(wb_mv1, Xcat, mv_b1, nullptr, hmid, 256, 0, 1, 0);
  xc_cm(hmid, Xa);
  gemm256(wb_mvs, Xcat, mv_bs, nullptr, val0, 256, 0, 0, 0);
  gemm128(wb_mv2, Xa, mv_b2, val0, val0, 128, 0, 0, 0);

  // projections: k = wk·key, q = wq·query, v = wv·val0
  gemm128(wb_wk, Xcat, bk, nullptr, kbuf, 256, 0, 0, 0);
  gemm128(wb_wq, Xcat, bq, nullptr, qbuf, 256, 128, 0, 0);
  xc_cm(val0, Xa);
  gemm128(wb_wv, Xa, bv, nullptr, vbuf, 128, 0, 0, 0);

  // ---- block 1: xyz-space ----
  knn_kernel<<<BN / 256, 256, 0, stream>>>(xyz, 3, idx1);
  emb_kernel<<<PAIRS / 256, 256, 0, stream>>>(xyz, 3, idx1, kbuf, qbuf, vbuf,
                                              xyz_w1, xyz_b1, xyz_w2, xyz_b2,
                                              Tbuf, EVbuf);
  attn_kernel<<<PAIRS / 128, 256, 128 * LDH * 2, stream>>>(
      Tbuf, EVbuf, wb_x1, xat_b1, wb_x2, xat_b2, aggb);

  // value = wkq·agg + bkq + identity ; value = wv·value + bv
  xc_cm(aggb, Xa);
  gemm128(wb_wkq, Xa, bkq, val0, tmp1, 128, 0, 0, 0);
  xc_cm(tmp1, Xa);
  gemm128(wb_wv, Xa, bv, nullptr, val2, 128, 0, 0, 0);

  // ---- block 2: pos-space (q/k swapped) ----
  knn_kernel<<<BN / 256, 256, 0, stream>>>(pos, 2, idx2);
  emb_kernel<<<PAIRS / 256, 256, 0, stream>>>(pos, 2, idx2, qbuf, kbuf, val2,
                                              pos_w1, pos_b1, pos_w2, pos_b2,
                                              Tbuf, EVbuf);
  attn_kernel<<<PAIRS / 128, 256, 128 * LDH * 2, stream>>>(
      Tbuf, EVbuf, wb_a1, at_b1, wb_a2, at_b2, aggb);

  // y = wend·agg + bend + identity  (row-major output)
  xc_cm(aggb, Xa);
  gemm128(wb_wend, Xa, bend, val0, out, 128, 0, 0, 1);
}
